// CausalSelfAttention_32847909880469
// MI455X (gfx1250) — matrix operand order repaired and checked by reading, not compiled, not run on hardware
//
#include <hip/hip_runtime.h>
#include <hip/hip_bf16.h>

// ---------------------------------------------------------------------------
// Problem constants (from the reference)
// ---------------------------------------------------------------------------
#define Bc      4
#define Tc      1000
#define Cc      1024
#define Hc      16
#define Dc      64
#define Mrows   (Bc * Tc)     // 4000, multiple of 16
#define TQP     1008          // queries padded to multiple of 16
#define TKP     1024          // keys/values padded to multiple of 32
#define NQT     (TQP / 16)    // 63 query tiles
#define BH      (Bc * Hc)     // 64

typedef __attribute__((ext_vector_type(16))) __bf16 v16bf;
typedef __attribute__((ext_vector_type(8)))  float  v8f;

union FragBF { unsigned u[8]; v16bf v; };

__device__ __forceinline__ unsigned short f2bf(float f) {
  unsigned x = __float_as_uint(f);
  unsigned r = x + 0x7FFFu + ((x >> 16) & 1u);   // round-to-nearest-even
  return (unsigned short)(r >> 16);
}
__device__ __forceinline__ unsigned pack_bf2(float a, float b) {
  return (unsigned)f2bf(a) | ((unsigned)f2bf(b) << 16);
}
__device__ __forceinline__ float halfmax(float v) {
  v = fmaxf(v, __shfl_xor(v, 1, 32));
  v = fmaxf(v, __shfl_xor(v, 2, 32));
  v = fmaxf(v, __shfl_xor(v, 4, 32));
  v = fmaxf(v, __shfl_xor(v, 8, 32));
  return v;
}
__device__ __forceinline__ float halfsum(float v) {
  v += __shfl_xor(v, 1, 32);
  v += __shfl_xor(v, 2, 32);
  v += __shfl_xor(v, 4, 32);
  v += __shfl_xor(v, 8, 32);
  return v;
}
// A-fragment K index for 16-bit 16x32 tiles (ISA 7.12.2): lane half + j
__device__ __forceinline__ int kA(int j, int half) {
  return ((j >> 2) << 4) + (half << 3) + ((j & 3) << 1);
}
// Workgroup-relative LDS byte offset of a shared-memory pointer.
// Flat aperture mapping (ISA 10.2): the LDS offset is the low 32 bits of the flat address.
__device__ __forceinline__ unsigned lds_off(const void* p) {
  return (unsigned)(unsigned long long)p;
}

// ---------------------------------------------------------------------------
// f32 -> bf16 converts (plain and transposed)
// ---------------------------------------------------------------------------
__global__ __launch_bounds__(256) void cvt_bf16(const float* __restrict__ in,
                                                unsigned short* __restrict__ out, int n) {
  int i = blockIdx.x * blockDim.x + threadIdx.x;
  if (i < n) out[i] = f2bf(in[i]);
}
// in[K][N] (fan-in major, as stored) -> out[N][K]
__global__ __launch_bounds__(256) void cvtT_bf16(const float* __restrict__ in,
                                                 unsigned short* __restrict__ out,
                                                 int K, int N) {
  int i = blockIdx.x * blockDim.x + threadIdx.x;
  if (i < K * N) {
    int k = i / N, n = i - k * N;
    out[(size_t)n * K + k] = f2bf(in[i]);
  }
}

// ---------------------------------------------------------------------------
// Block-cooperative bf16 WMMA GEMM: C[M,N] = A[M,K] * B (Bt stored [N,K]).
// Block = 8 waves = 128 rows x 64 cols. Per K=32 step the 64x32 bf16 B tile
// (4KB) is staged into LDS with ONE global_load_async_to_lds_b128 per thread
// (256 threads x 16B), double-buffered on ASYNCcnt, shared by all 8 waves.
// A fragments are per-wave contiguous global b128 loads.
// ---------------------------------------------------------------------------
__global__ __launch_bounds__(256) void gemm_bf16(const unsigned short* __restrict__ A,
                                                 const unsigned short* __restrict__ Bt,
                                                 float* __restrict__ C,
                                                 int M, int N, int K) {
  __shared__ __align__(16) unsigned short ldsB[2][64][32];  // [buf][n][k] bf16
  int tid  = threadIdx.x;
  int wid  = tid >> 5;
  int lane = tid & 31;
  int tilesN = N >> 6;
  int bm = blockIdx.x / tilesN;          // 128-row block
  int tn = blockIdx.x - bm * tilesN;     // 64-col tile
  int tm = bm * 8 + wid;                 // this wave's 16-row tile
  int half = lane >> 4, nl = lane & 15;

  int rowA = tm * 16 + nl;
  if (rowA > M - 1) rowA = M - 1;        // clamp (stores are guarded)
  const unsigned* Arow = (const unsigned*)(A + (size_t)rowA * K);

  // async B-tile staging: thread copies 16 bytes of the 64x32 tile
  int brow = tid >> 2;                                   // n within tile
  unsigned voffBase = (unsigned)(((tn * 64 + brow) * K + (tid & 3) * 8) * 2);
  unsigned long long bbase = (unsigned long long)Bt;
  unsigned ldsDst0 = lds_off(&ldsB[0][0][0]) + (unsigned)(tid * 16);

  v8f acc[4] = {v8f(0.f), v8f(0.f), v8f(0.f), v8f(0.f)};
  int nsteps = K >> 5;

  // preload tile 0
  {
    unsigned d = ldsDst0;
    unsigned v = voffBase;
    asm volatile("global_load_async_to_lds_b128 %0, %1, %2"
                 :: "v"(d), "v"(v), "s"(bbase) : "memory");
  }
  for (int i = 0; i < nsteps; i++) {
    int cur = i & 1;
    if (i + 1 < nsteps) {
      unsigned d = ldsDst0 + (unsigned)((cur ^ 1) * (int)sizeof(ldsB[0]));
      unsigned v = voffBase + (unsigned)((i + 1) * 64);  // 32 bf16 = 64B per step
      asm volatile("global_load_async_to_lds_b128 %0, %1, %2"
                   :: "v"(d), "v"(v), "s"(bbase) : "memory");
      asm volatile("s_wait_asynccnt 0x1" ::: "memory");  // oldest (cur tile) done
    } else {
      asm volatile("s_wait_asynccnt 0x0" ::: "memory");
    }
    __syncthreads();                                     // whole cur tile visible

    FragBF a;
    int k0 = i * 32;
#pragma unroll
    for (int j = 0; j < 8; j++) a.u[j] = Arow[(k0 + half * 16 + 2 * j) >> 1];
#pragma unroll
    for (int nf = 0; nf < 4; nf++) {
      const unsigned* lb = (const unsigned*)&ldsB[cur][nf * 16 + nl][0];
      FragBF b;
#pragma unroll
      for (int j = 0; j < 8; j++) b.u[j] = lb[half * 8 + j];  // contiguous b128s
      acc[nf] = __builtin_amdgcn_wmma_f32_16x16x32_bf16(
          false, a.v, false, b.v, (short)0, acc[nf], false, false);
    }
    __syncthreads();   // protect buffer about to be overwritten by next prefetch
  }

  if (tm * 16 < M) {   // M is a multiple of 16, so whole tile is in-range
#pragma unroll
    for (int nf = 0; nf < 4; nf++)
#pragma unroll
      for (int r = 0; r < 8; r++)
        C[(size_t)(tm * 16 + r + half * 8) * N + tn * 64 + nf * 16 + nl] = acc[nf][r];
  }
}

// ---------------------------------------------------------------------------
// RoPE + split: qkv fp32 [B*T, 3C] -> Qb [BH,TQP,64] bf16 (roped),
//               Kb [BH,TKP,64] bf16 (roped), Vt [BH,64,TKP] bf16 (transposed)
// Padded t writes zeros.
// ---------------------------------------------------------------------------
__global__ __launch_bounds__(256) void rope_split(const float* __restrict__ qkv,
                                                  unsigned short* __restrict__ Qb,
                                                  unsigned short* __restrict__ Kb,
                                                  unsigned short* __restrict__ Vt) {
  int idx = blockIdx.x * blockDim.x + threadIdx.x;
  int i  = idx & 31;           // rotation pair index (d = 2i, 2i+1)
  int t  = (idx >> 5) & 1023;  // 0..TKP-1
  int bh = idx >> 15;
  if (bh >= BH) return;
  int h = bh & 15, b = bh >> 4;

  float q0 = 0.f, q1 = 0.f, k0 = 0.f, k1 = 0.f, v0 = 0.f, v1 = 0.f;
  if (t < Tc) {
    const float* row = qkv + (size_t)(b * Tc + t) * (3 * Cc);
    int c = h * Dc + 2 * i;
    q0 = row[c];          q1 = row[c + 1];
    k0 = row[Cc + c];     k1 = row[Cc + c + 1];
    v0 = row[2 * Cc + c]; v1 = row[2 * Cc + c + 1];
    float theta = __expf(-(float)(2 * i) * (9.210340372f / 64.0f));  // 10000^(-2i/64)
    float ang = (float)t * theta;
    float sn, cs;
    sincosf(ang, &sn, &cs);
    float a0 = q0 * cs - q1 * sn, a1 = q1 * cs + q0 * sn;
    float b0 = k0 * cs - k1 * sn, b1 = k1 * cs + k0 * sn;
    q0 = a0; q1 = a1; k0 = b0; k1 = b1;
  }
  unsigned short* kp = Kb + ((size_t)bh * TKP + t) * Dc + 2 * i;
  kp[0] = f2bf(k0); kp[1] = f2bf(k1);
  Vt[((size_t)bh * Dc + 2 * i    ) * TKP + t] = f2bf(v0);
  Vt[((size_t)bh * Dc + 2 * i + 1) * TKP + t] = f2bf(v1);
  if (t < TQP) {
    unsigned short* qp = Qb + ((size_t)bh * TQP + t) * Dc + 2 * i;
    qp[0] = f2bf(q0); qp[1] = f2bf(q1);
  }
}

// ---------------------------------------------------------------------------
// Flash attention: one wave per (b, h, 16-query tile).
// Per 32-key tile: S(16x32) via 4 WMMAs, scale + causal mask, online softmax,
// P re-laid out via per-wave LDS slab (C-frag -> A-frag), O += P*V (4 WMMAs).
// Output written directly as bf16 y[B,T,H,D] for the projection GEMM.
// ---------------------------------------------------------------------------
__global__ __launch_bounds__(256) void attn_flash(const unsigned short* __restrict__ Qb,
                                                  const unsigned short* __restrict__ Kb,
                                                  const unsigned short* __restrict__ Vt,
                                                  unsigned short* __restrict__ yb) {
  __shared__ float ldsP[8][16][32];   // per-wave 2KB staging slab
  int wid  = threadIdx.x >> 5;
  int lane = threadIdx.x & 31;
  int gw = blockIdx.x * 8 + wid;
  if (gw >= BH * NQT) return;
  int bh = gw / NQT, qt = gw - bh * NQT;
  int b = bh >> 4, h = bh & 15;
  int half = lane >> 4, nl = lane & 15;

  // Q fragments for the 16 query rows (two K=32 chunks over D=64)
  FragBF aQ[2];
  {
    const unsigned* qrow = (const unsigned*)(Qb + ((size_t)bh * TQP + qt * 16 + nl) * Dc);
#pragma unroll
    for (int c = 0; c < 2; c++)
#pragma unroll
      for (int j = 0; j < 8; j++) aQ[c].u[j] = qrow[(c * 32 + half * 16 + 2 * j) >> 1];
  }

  v8f O[4] = {v8f(0.f), v8f(0.f), v8f(0.f), v8f(0.f)};
  float mrow[8], lrow[8];
#pragma unroll
  for (int r = 0; r < 8; r++) { mrow[r] = -INFINITY; lrow[r] = 0.f; }

  const float scale = 0.125f;  // 1/sqrt(64)
  int nkt = (qt * 16 + 15) / 32 + 1;

  for (int kt = 0; kt < nkt; kt++) {
    float S[2][8];
#pragma unroll
    for (int s = 0; s < 2; s++) {
      const unsigned* krow =
          (const unsigned*)(Kb + ((size_t)bh * TKP + kt * 32 + s * 16 + nl) * Dc);
      FragBF bK0, bK1;
#pragma unroll
      for (int j = 0; j < 8; j++) {
        int d = half * 16 + 2 * j;
        bK0.u[j] = krow[d >> 1];
        bK1.u[j] = krow[(32 + d) >> 1];
      }
      v8f sacc = v8f(0.f);
      sacc = __builtin_amdgcn_wmma_f32_16x16x32_bf16(false, aQ[0].v, false, bK0.v,
                                                     (short)0, sacc, false, false);
      sacc = __builtin_amdgcn_wmma_f32_16x16x32_bf16(false, aQ[1].v, false, bK1.v,
                                                     (short)0, sacc, false, false);
#pragma unroll
      for (int r = 0; r < 8; r++) {
        int q  = qt * 16 + r + half * 8;
        int kk = kt * 32 + s * 16 + nl;
        S[s][r] = (kk <= q) ? sacc[r] * scale : -INFINITY;
      }
    }
    // online softmax; row stats replicated across each 16-lane half
#pragma unroll
    for (int r = 0; r < 8; r++) {
      float rm   = halfmax(fmaxf(S[0][r], S[1][r]));
      float mnew = fmaxf(mrow[r], rm);
      float alpha = __expf(mrow[r] - mnew);       // exp(-inf - finite) = 0 first pass
      float p0 = __expf(S[0][r] - mnew);
      float p1 = __expf(S[1][r] - mnew);
      S[0][r] = p0; S[1][r] = p1;
      float rs = halfsum(p0 + p1);
      lrow[r] = lrow[r] * alpha + rs;
      mrow[r] = mnew;
#pragma unroll
      for (int dt = 0; dt < 4; dt++) O[dt][r] *= alpha;
    }
    // C-frag -> A-frag re-layout of P through per-wave LDS (wave-local, DS is
    // in-order per wave; compiler inserts s_wait_dscnt for the dependent loads)
#pragma unroll
    for (int s = 0; s < 2; s++)
#pragma unroll
      for (int r = 0; r < 8; r++) ldsP[wid][r + half * 8][s * 16 + nl] = S[s][r];
    FragBF aP;
#pragma unroll
    for (int j = 0; j < 8; j++) {
      int k = half * 16 + 2 * j;
      aP.u[j] = pack_bf2(ldsP[wid][nl][k], ldsP[wid][nl][k + 1]);
    }
    // O += P(16x32) * V(32x64): V transposed so key-pairs are contiguous
#pragma unroll
    for (int dt = 0; dt < 4; dt++) {
      const unsigned* vrow =
          (const unsigned*)(Vt + ((size_t)bh * Dc + dt * 16 + nl) * TKP);
      FragBF bV;
#pragma unroll
      for (int j = 0; j < 8; j++) bV.u[j] = vrow[(kt * 32 + half * 16 + 2 * j) >> 1];
      O[dt] = __builtin_amdgcn_wmma_f32_16x16x32_bf16(false, aP.v, false, bV.v,
                                                      (short)0, O[dt], false, false);
    }
  }

  // epilogue: normalize and emit bf16 y[B,T,H,D] (== [B*T, C] rows)
#pragma unroll
  for (int dt = 0; dt < 4; dt++)
#pragma unroll
    for (int r = 0; r < 8; r++) {
      int t = qt * 16 + r + half * 8;
      if (t < Tc) {
        float val = O[dt][r] / lrow[r];
        yb[(size_t)(b * Tc + t) * Cc + h * Dc + dt * 16 + nl] = f2bf(val);
      }
    }
}

// ---------------------------------------------------------------------------
// Host-side launcher
// ---------------------------------------------------------------------------
static inline size_t alignUp(size_t x) { return (x + 255) & ~(size_t)255; }

extern "C" void kernel_launch(void* const* d_in, const int* in_sizes, int n_in,
                              void* d_out, int out_size, void* d_ws, size_t ws_size,
                              hipStream_t stream) {
  (void)in_sizes; (void)n_in; (void)out_size; (void)ws_size;
  const float* x      = (const float*)d_in[0];   // [4,1000,1024]
  // d_in[1] = attn_mask (causal) -- implied by kernel, unused
  const float* W_attn = (const float*)d_in[2];   // [1024, 3072]
  const float* W_proj = (const float*)d_in[3];   // [1024, 1024]
  float* out = (float*)d_out;                    // [4,1000,1024] fp32

  // workspace partition
  char* p = (char*)d_ws;
  size_t off = 0;
  unsigned short* xb   = (unsigned short*)(p + off); off = alignUp(off + (size_t)Mrows * Cc * 2);
  unsigned short* WaT  = (unsigned short*)(p + off); off = alignUp(off + (size_t)3 * Cc * Cc * 2);
  unsigned short* WpT  = (unsigned short*)(p + off); off = alignUp(off + (size_t)Cc * Cc * 2);
  float*          qkv  = (float*)(p + off);          off = alignUp(off + (size_t)Mrows * 3 * Cc * 4);
  unsigned short* Qb   = (unsigned short*)(p + off); off = alignUp(off + (size_t)BH * TQP * Dc * 2);
  unsigned short* Kb   = (unsigned short*)(p + off); off = alignUp(off + (size_t)BH * TKP * Dc * 2);
  unsigned short* Vt   = (unsigned short*)(p + off); off = alignUp(off + (size_t)BH * Dc * TKP * 2);
  unsigned short* ybuf = (unsigned short*)(p + off); off = alignUp(off + (size_t)Mrows * Cc * 2 + 65536);

  const int TB = 256;
  // 1) converts
  {
    int n = Mrows * Cc;
    cvt_bf16<<<(n + TB - 1) / TB, TB, 0, stream>>>(x, xb, n);
  }
  {
    int n = Cc * 3 * Cc;
    cvtT_bf16<<<(n + TB - 1) / TB, TB, 0, stream>>>(W_attn, WaT, Cc, 3 * Cc);
  }
  {
    int n = Cc * Cc;
    cvtT_bf16<<<(n + TB - 1) / TB, TB, 0, stream>>>(W_proj, WpT, Cc, Cc);
  }
  // 2) qkv = x @ W_attn   (M=4000, N=3072, K=1024); 128-row blocks
  {
    int blocks = ((Mrows + 127) / 128) * (3 * Cc / 64);   // 32*48
    gemm_bf16<<<blocks, TB, 0, stream>>>(xb, WaT, qkv, Mrows, 3 * Cc, Cc);
  }
  // 3) RoPE + split (+pad, +V transpose)
  {
    int n = BH * TKP * (Dc / 2);                          // 2,097,152
    rope_split<<<n / TB, TB, 0, stream>>>(qkv, Qb, Kb, Vt);
  }
  // 4) flash attention -> yb (bf16)
  {
    int waves = BH * NQT;                                 // 4032
    attn_flash<<<(waves + 7) / 8, TB, 0, stream>>>(Qb, Kb, Vt, ybuf);
  }
  // 5) out = y @ W_proj  (M=4000, N=1024, K=1024)
  {
    int blocks = ((Mrows + 127) / 128) * (Cc / 64);       // 32*16
    gemm_bf16<<<blocks, TB, 0, stream>>>(ybuf, WpT, out, Mrows, Cc, Cc);
  }
}
